// Zero_Parameter_67834713473347
// MI455X (gfx1250) — matrix operand order repaired and checked by reading, not compiled, not run on hardware
//
#include <hip/hip_runtime.h>
#include <hip/hip_bf16.h>
#include <math.h>

typedef __bf16 bf16;
typedef __attribute__((ext_vector_type(16))) __bf16 v16bf;
typedef __attribute__((ext_vector_type(8)))  __bf16 v8bf;
typedef __attribute__((ext_vector_type(8)))  float  v8f;

#define SEQ     2048
#define DMODEL  768
#define HEADS   8
#define HD      96
#define BATCH   4
#define ATTN_SCALE 0.036084391824351614f   // 768^-0.5

// A-operand (16-bit 16x32): per lane (hi = lane>>4) the 16 elements
// are one contiguous 16-element run: k = hi*16 + 0..15 (matches B operands).
__device__ __forceinline__ v16bf load_a_operand(const bf16* rowptr, int hi) {
    v8bf lo = *reinterpret_cast<const v8bf*>(rowptr + hi * 16);
    v8bf hh = *reinterpret_cast<const v8bf*>(rowptr + hi * 16 + 8);
    return __builtin_shufflevector(lo, hh, 0,1,2,3,4,5,6,7,8,9,10,11,12,13,14,15);
}

__device__ __forceinline__ float rowsum16(float v) {
    #pragma unroll
    for (int m = 1; m < 16; m <<= 1) v += __shfl_xor(v, m, 32);
    return v;
}

// ---------------------------------------------------------------------------
// GEMM: out[row0:+16, col0:+64] = A[M,KDIM] @ B[KDIM,768] (+bias)
// 128 threads = 4 waves; wave wv owns the 16x16 tile at column wv*16.
// K unrolled by 64: two WMMAs per barrier stage. B staged transposed so the
// B operand is one contiguous 32B LDS vector load.
// ---------------------------------------------------------------------------
template <int KDIM, bool WITH_BIAS>
__global__ __launch_bounds__(128)
void proj_gemm_kernel(const float* __restrict__ A,
                      const float* __restrict__ Bm,
                      const float* __restrict__ bias,
                      const int* __restrict__ layer_ptr,
                      float* __restrict__ out_f32,
                      bf16* __restrict__ out_bf) {
    __shared__ __attribute__((aligned(32))) bf16 Alds[16][64];   // M x K     (2 KB)
    __shared__ __attribute__((aligned(32))) bf16 BT[64][64];     // N x K (T) (8 KB)

    const int tid  = threadIdx.x;
    const int lane = tid & 31;
    const int wv   = tid >> 5;
    const int n    = lane & 15;
    const int hi   = lane >> 4;
    const int row0 = blockIdx.x * 16;
    const int col0 = blockIdx.y * 64;

    if (layer_ptr) Bm += (size_t)(*layer_ptr) * KDIM * DMODEL;

    v8f acc = {0.f, 0.f, 0.f, 0.f, 0.f, 0.f, 0.f, 0.f};

    for (int k0 = 0; k0 < KDIM; k0 += 64) {
        for (int i = tid; i < 16 * 64; i += 128) {
            int r = i >> 6, c = i & 63;
            Alds[r][c] = (bf16)A[(size_t)(row0 + r) * KDIM + (k0 + c)];
        }
        for (int i = tid; i < 64 * 64; i += 128) {
            int r = i >> 6, c = i & 63;                 // coalesced global read
            BT[c][r] = (bf16)Bm[(size_t)(k0 + r) * DMODEL + (col0 + c)];
        }
        __syncthreads();

        #pragma unroll
        for (int kk = 0; kk < 64; kk += 32) {
            v16bf av = load_a_operand(&Alds[n][kk], hi);
            v16bf bv = *reinterpret_cast<const v16bf*>(&BT[wv * 16 + n][kk + hi * 16]);
            acc = __builtin_amdgcn_wmma_f32_16x16x32_bf16(false, av, false, bv,
                                                          (short)0, acc, false, false);
        }
        __syncthreads();
    }

    #pragma unroll
    for (int r = 0; r < 8; ++r) {
        const int m   = r + 8 * hi;
        const int col = col0 + wv * 16 + n;
        const size_t idx = (size_t)(row0 + m) * DMODEL + col;
        if (WITH_BIAS) out_bf[idx]  = (bf16)(acc[r] + bias[col]);
        else           out_f32[idx] = acc[r];
    }
}

// ---------------------------------------------------------------------------
// Row L2-normalize F_s_p in place (fp32) + bf16 copy for attention.
// ---------------------------------------------------------------------------
__global__ __launch_bounds__(256)
void norm_kv_kernel(float* __restrict__ fsp, bf16* __restrict__ kv_bf) {
    __shared__ float red[256];
    const int row = blockIdx.x;
    const int tid = threadIdx.x;
    float vals[3], ss = 0.f;
    #pragma unroll
    for (int i = 0; i < 3; ++i) {
        float v = fsp[(size_t)row * DMODEL + tid + 256 * i];
        vals[i] = v; ss += v * v;
    }
    red[tid] = ss; __syncthreads();
    for (int s = 128; s > 0; s >>= 1) {
        if (tid < s) red[tid] += red[tid + s];
        __syncthreads();
    }
    const float inv = 1.0f / sqrtf(red[0]);
    #pragma unroll
    for (int i = 0; i < 3; ++i) {
        const size_t idx = (size_t)row * DMODEL + tid + 256 * i;
        float nv = vals[i] * inv;
        fsp[idx]   = nv;
        kv_bf[idx] = (bf16)nv;
    }
}

// ---------------------------------------------------------------------------
// Attention. grid=(N1/64, B*H), block=128 (4 waves); wave = 16 query rows.
// kv rows are unit-norm, so SCALE*||q_row|| statically upper-bounds every
// logit -> fixed softmax shift, no online max/rescale; row-sum kept as
// per-lane partials, reduced once after the loop. S tiles processed
// sequentially to minimize live registers.
// ---------------------------------------------------------------------------
__global__ __launch_bounds__(128) __attribute__((amdgpu_waves_per_eu(1, 4)))
void attn_kernel(const bf16* __restrict__ qb,
                 const bf16* __restrict__ kvb,
                 float* __restrict__ out) {
    __shared__ __attribute__((aligned(32))) bf16 kvs[32][HD];     // row-major, S-phase
    __shared__ __attribute__((aligned(32))) bf16 kvT[HD][32];     // transposed, PV-phase
    __shared__ __attribute__((aligned(32))) bf16 pls[4][16][32];  // P staging per wave
    __shared__ float qss[4][32];

    const int tid  = threadIdx.x;
    const int lane = tid & 31;
    const int wv   = tid >> 5;
    const int n    = lane & 15;
    const int hi   = lane >> 4;
    const int bh   = blockIdx.y;
    const int b    = bh >> 3;
    const int h    = bh & 7;
    const int qrow0 = blockIdx.x * 64 + wv * 16;

    // Q tile (16 x 96) as three A operands, each = two 16B global loads.
    const size_t qbase = ((size_t)b * SEQ + qrow0) * DMODEL + h * HD;
    v16bf aq[3];
    #pragma unroll
    for (int c = 0; c < 3; ++c)
        aq[c] = load_a_operand(qb + qbase + (size_t)n * DMODEL + c * 32, hi);

    // per-lane partial ||q_row||^2 (row m = n; lanes n and n+16 hold halves)
    float ssq = 0.f;
    #pragma unroll
    for (int c = 0; c < 3; ++c)
        #pragma unroll
        for (int j = 0; j < 16; ++j) { float qv = (float)aq[c][j]; ssq += qv * qv; }
    qss[wv][lane] = ssq;
    asm volatile("s_wait_dscnt 0x0" ::: "memory");
    float bound[8];
    #pragma unroll
    for (int r = 0; r < 8; ++r) {
        const int m = r + 8 * hi;                       // D-layout row for this lane
        bound[r] = ATTN_SCALE * sqrtf(qss[wv][m] + qss[wv][16 + m]);
    }

    v8f o[6];
    float rsum[8];
    #pragma unroll
    for (int c6 = 0; c6 < 6; ++c6)
        #pragma unroll
        for (int r = 0; r < 8; ++r) o[c6][r] = 0.f;
    #pragma unroll
    for (int r = 0; r < 8; ++r) rsum[r] = 0.f;

    const size_t kvbase = (size_t)b * SEQ * DMODEL + h * HD;

    for (int m0 = 0; m0 < SEQ; m0 += 32) {
        __syncthreads();                               // protect kvs/kvT reuse
        for (int i = tid; i < 32 * HD; i += 128) {
            int kk = i / HD, dd = i - kk * HD;
            bf16 v = kvb[kvbase + (size_t)(m0 + kk) * DMODEL + dd];
            kvs[kk][dd] = v;
            kvT[dd][kk] = v;
        }
        __syncthreads();

        // S tiles sequentially (16 kv rows each) to keep live ranges short
        #pragma unroll
        for (int t = 0; t < 2; ++t) {
            v8f s = {0.f,0.f,0.f,0.f,0.f,0.f,0.f,0.f};
            #pragma unroll
            for (int c = 0; c < 3; ++c) {
                v16bf bK = *reinterpret_cast<const v16bf*>(
                    &kvs[t * 16 + n][c * 32 + hi * 16]);
                s = __builtin_amdgcn_wmma_f32_16x16x32_bf16(false, aq[c], false, bK,
                                                            (short)0, s, false, false);
            }
            #pragma unroll
            for (int r = 0; r < 8; ++r) {
                const float p = __expf(s[r] * ATTN_SCALE - bound[r]);
                rsum[r] += p;
                pls[wv][r + 8 * hi][t * 16 + n] = (bf16)p;
            }
        }
        asm volatile("s_wait_dscnt 0x0" ::: "memory");  // same-wave LDS RAW

        // P as A operand (16 q x 32 kv): two contiguous 16B LDS loads
        v16bf ap = load_a_operand(&pls[wv][n][0], hi);

        // O += P @ KV  (six 16-dim chunks; B operand = one 32B LDS vector load)
        #pragma unroll
        for (int c6 = 0; c6 < 6; ++c6) {
            v16bf bv = *reinterpret_cast<const v16bf*>(&kvT[c6 * 16 + n][hi * 16]);
            o[c6] = __builtin_amdgcn_wmma_f32_16x16x32_bf16(false, ap, false, bv,
                                                            (short)0, o[c6], false, false);
        }
    }

    // single post-loop cross-lane reduction of row sums
    #pragma unroll
    for (int r = 0; r < 8; ++r) {
        const float inv = 1.0f / rowsum16(rsum[r]);
        const size_t row = (size_t)b * SEQ + qrow0 + r + 8 * hi;
        #pragma unroll
        for (int c6 = 0; c6 < 6; ++c6)
            out[row * DMODEL + h * HD + c6 * 16 + n] = o[c6][r] * inv;
    }
}

// ---------------------------------------------------------------------------
// out_ft = L2norm(out_ft + F_t) per row of 768.
// ---------------------------------------------------------------------------
__global__ __launch_bounds__(256)
void finalize_kernel(float* __restrict__ out_ft, const float* __restrict__ F_t) {
    __shared__ float red[256];
    const int row = blockIdx.x;
    const int tid = threadIdx.x;
    float vals[3], ss = 0.f;
    #pragma unroll
    for (int i = 0; i < 3; ++i) {
        const size_t idx = (size_t)row * DMODEL + tid + 256 * i;
        float v = out_ft[idx] + F_t[idx];
        vals[i] = v; ss += v * v;
    }
    red[tid] = ss; __syncthreads();
    for (int s = 128; s > 0; s >>= 1) {
        if (tid < s) red[tid] += red[tid + s];
        __syncthreads();
    }
    const float inv = 1.0f / sqrtf(red[0]);
    #pragma unroll
    for (int i = 0; i < 3; ++i)
        out_ft[(size_t)row * DMODEL + tid + 256 * i] = vals[i] * inv;
}

// ---------------------------------------------------------------------------
extern "C" void kernel_launch(void* const* d_in, const int* in_sizes, int n_in,
                              void* d_out, int out_size, void* d_ws, size_t ws_size,
                              hipStream_t stream) {
    const float* F_t   = (const float*)d_in[0];
    const float* F_s   = (const float*)d_in[1];
    const float* W_lin = (const float*)d_in[2];
    const float* W_q   = (const float*)d_in[3];
    const float* b_q   = (const float*)d_in[4];
    const int*   layer = (const int*)d_in[5];

    const size_t HALF = (size_t)BATCH * SEQ * DMODEL;   // 6,291,456
    float* out_ft  = (float*)d_out;
    float* out_fsp = out_ft + HALF;

    bf16* q_bf  = (bf16*)d_ws;          // 12.6 MB
    bf16* kv_bf = q_bf + HALF;          // 12.6 MB

    const dim3 gemm_grid((BATCH * SEQ) / 16, DMODEL / 64);  // (512, 12)

    proj_gemm_kernel<1024, false><<<gemm_grid, 128, 0, stream>>>(
        F_s, W_lin, nullptr, layer, out_fsp, nullptr);

    norm_kv_kernel<<<BATCH * SEQ, 256, 0, stream>>>(out_fsp, kv_bf);

    proj_gemm_kernel<768, true><<<gemm_grid, 128, 0, stream>>>(
        F_t, W_q, b_q, nullptr, nullptr, q_bf);

    attn_kernel<<<dim3(SEQ / 64, BATCH * HEADS), 128, 0, stream>>>(q_bf, kv_bf, out_ft);

    finalize_kernel<<<BATCH * SEQ, 256, 0, stream>>>(out_ft, F_t);
}
